// Seq2Seq_68264210203039
// MI455X (gfx1250) — matrix operand order repaired and checked by reading, not compiled, not run on hardware
//
#include <hip/hip_runtime.h>
#include <hip/hip_bf16.h>

// Problem constants (from reference)
#define Bc 4
#define BEAMc 5
#define TMAXc 16
#define Vc 50257
#define Dc 768
#define Hc 12
#define NLc 2
#define Sc 128
#define DFFc 3072
#define BUFc (TMAXc + 1)      // 17
#define DHc (Dc / Hc)         // 64
#define BQc (Bc * BEAMc)      // 20
#define ROWSc (BQc * BUFc)    // 340
#define SOSc 1
#define EOSc 2
#define NEGc (-1e20f)

typedef __bf16 bf16_t;
typedef bf16_t v16bf __attribute__((ext_vector_type(16)));
typedef float  v8f   __attribute__((ext_vector_type(8)));
typedef float  v4f   __attribute__((ext_vector_type(4)));
typedef unsigned short v8u16 __attribute__((ext_vector_type(8)));

__device__ __forceinline__ unsigned short f2bfu(float f) {
  unsigned int u = __builtin_bit_cast(unsigned int, f);
  return (unsigned short)((u + 0x7FFFu + ((u >> 16) & 1u)) >> 16);
}
__device__ __forceinline__ bf16_t u2bf(unsigned short h) {
  return __builtin_bit_cast(bf16_t, h);
}

// ---------------------------------------------------------------------------
// WMMA GEMM: C[M,N] = act(A[M,K] @ W[N,K]^T + bias[N])
// ABF/BBF: 0 = operand is f32 (converted to bf16 on the fly), 1 = raw bf16
// ACT:     0 = none, 1 = relu, 2 = tanh
// One wave computes one 16x16 tile via v_wmma_f32_16x16x32_bf16, K step 32.
// OOB rows/cols are handled by CLAMPING the source pointer (their products
// land only in discarded output rows/cols), so all loads are unpredicated
// contiguous vector loads:
//   A lane (m = lane&15, half = lane>>4): elems 0-7 = A[m][kk+half*8 .. +8),
//                                         elems 8-15 = A[m][kk+16+half*8 .. +8)
//   B lane (n = lane&15, half = lane>>4): elems 0-7 = W[n][kk+half*8 .. +8),
//                                         elems 8-15 = W[n][kk+16+half*8 .. +8)
//   C/D: elem j -> row m0 + j + half*8, col n0 + (lane&15)
// ---------------------------------------------------------------------------
template <int ABF, int BBF, int ACT>
__global__ __launch_bounds__(32) void gemm_wmma(const void* __restrict__ Av,
                                                const void* __restrict__ Wv,
                                                const float* __restrict__ bias,
                                                float* __restrict__ C,
                                                int M, int N, int K) {
  const float* Af = (const float*)Av;
  const unsigned short* Ah = (const unsigned short*)Av;
  const float* Wf = (const float*)Wv;
  const unsigned short* Wh = (const unsigned short*)Wv;
  const int lane = threadIdx.x & 31;
  const int half = lane >> 4;
  const int mn   = lane & 15;
  const int m0 = blockIdx.y << 4;
  const int n0 = blockIdx.x << 4;
  const int arow = m0 + mn;
  const int bcol = n0 + mn;
  const int arowc = (arow < M) ? arow : (M - 1);
  const int bcolc = (bcol < N) ? bcol : (N - 1);
  const float* Afr = Af + (size_t)arowc * K + (half << 3);          // + half*8
  const unsigned short* Ahr = Ah + (size_t)arowc * K + (half << 3); // + half*8
  const float* Wfr = Wf + (size_t)bcolc * K + (half << 3);          // + half*8
  const unsigned short* Whr = Wh + (size_t)bcolc * K + (half << 3); // + half*8

  v8f acc = {0.f, 0.f, 0.f, 0.f, 0.f, 0.f, 0.f, 0.f};
  for (int kk = 0; kk < K; kk += 32) {
    v16bf av;
    if (ABF == 0) {
      // A f32: two contiguous 8-float runs -> 4 x b128 loads + convert
      const v4f a0 = *(const v4f*)(Afr + kk);
      const v4f a1 = *(const v4f*)(Afr + kk + 4);
      const v4f a2 = *(const v4f*)(Afr + kk + 16);
      const v4f a3 = *(const v4f*)(Afr + kk + 20);
#pragma unroll
      for (int i = 0; i < 4; ++i) {
        av[i]      = u2bf(f2bfu(a0[i]));
        av[i + 4]  = u2bf(f2bfu(a1[i]));
        av[i + 8]  = u2bf(f2bfu(a2[i]));
        av[i + 12] = u2bf(f2bfu(a3[i]));
      }
    } else {
      // A bf16: two contiguous 16-byte runs -> 2 x b128 loads, pure bitcast
      union { v8u16 h[2]; v16bf v; } ua;
      ua.h[0] = *(const v8u16*)(Ahr + kk);
      ua.h[1] = *(const v8u16*)(Ahr + kk + 16);
      av = ua.v;
    }
    v16bf bv;
    if (BBF == 0) {
      // W f32: two contiguous 8-float runs -> 4 x b128 loads + convert
      const v4f b0 = *(const v4f*)(Wfr + kk);
      const v4f b1 = *(const v4f*)(Wfr + kk + 4);
      const v4f b2 = *(const v4f*)(Wfr + kk + 16);
      const v4f b3 = *(const v4f*)(Wfr + kk + 20);
#pragma unroll
      for (int i = 0; i < 4; ++i) {
        bv[i]      = u2bf(f2bfu(b0[i]));
        bv[i + 4]  = u2bf(f2bfu(b1[i]));
        bv[i + 8]  = u2bf(f2bfu(b2[i]));
        bv[i + 12] = u2bf(f2bfu(b3[i]));
      }
    } else {
      // W bf16: two contiguous 16-byte runs -> 2 x b128 loads, pure bitcast
      union { v8u16 h[2]; v16bf v; } ub;
      ub.h[0] = *(const v8u16*)(Whr + kk);
      ub.h[1] = *(const v8u16*)(Whr + kk + 16);
      bv = ub.v;
    }
    acc = __builtin_amdgcn_wmma_f32_16x16x32_bf16(false, av, false, bv,
                                                  (short)0, acc, false, false);
  }
  const float bb = (bias != nullptr) ? bias[bcolc] : 0.f;
  if (m0 + 16 <= M && n0 + 16 <= N) {
    // interior tile: unconditional stores (uniform branch, no exec juggling)
#pragma unroll
    for (int j = 0; j < 8; ++j) {
      const int row = m0 + j + (half << 3);
      float v = acc[j] + bb;
      if (ACT == 1) v = fmaxf(v, 0.f);
      if (ACT == 2) v = tanhf(v);
      C[(size_t)row * N + bcol] = v;
    }
  } else {
#pragma unroll
    for (int j = 0; j < 8; ++j) {
      const int row = m0 + j + (half << 3);
      if (row < M && bcol < N) {
        float v = acc[j] + bb;
        if (ACT == 1) v = fmaxf(v, 0.f);
        if (ACT == 2) v = tanhf(v);
        C[(size_t)row * N + bcol] = v;
      }
    }
  }
}

// x[row,:] = LN(emb[tok] + pos[p]) * g + b      (one block per token row)
__global__ __launch_bounds__(256) void embed_ln(const float* __restrict__ emb,
                                                const float* __restrict__ pos,
                                                const float* __restrict__ g,
                                                const float* __restrict__ bta,
                                                const int* __restrict__ buf,
                                                float* __restrict__ x) {
  const int row = blockIdx.x;           // 0..ROWSc-1
  const int pidx = row % BUFc;
  const int tid = threadIdx.x;
  const int tok = buf[row];
  __shared__ float red[256];
  float v[3];
  float s = 0.f;
#pragma unroll
  for (int j = 0; j < 3; ++j) {
    const int i = tid + (j << 8);
    v[j] = emb[(size_t)tok * Dc + i] + pos[(size_t)pidx * Dc + i];
    s += v[j];
  }
  red[tid] = s; __syncthreads();
  for (int st = 128; st > 0; st >>= 1) { if (tid < st) red[tid] += red[tid + st]; __syncthreads(); }
  const float mean = red[0] * (1.f / Dc);
  __syncthreads();
  float s2 = 0.f;
#pragma unroll
  for (int j = 0; j < 3; ++j) { const float d = v[j] - mean; s2 += d * d; }
  red[tid] = s2; __syncthreads();
  for (int st = 128; st > 0; st >>= 1) { if (tid < st) red[tid] += red[tid + st]; __syncthreads(); }
  const float inv = rsqrtf(red[0] * (1.f / Dc) + 1e-5f);
#pragma unroll
  for (int j = 0; j < 3; ++j) {
    const int i = tid + (j << 8);
    x[(size_t)row * Dc + i] = (v[j] - mean) * inv * g[i] + bta[i];
  }
}

// o[row,:] = LN(x[row,:] + y[row,:]) * g + b   (in-place safe: no restrict on x/o)
__global__ __launch_bounds__(256) void add_ln(const float* x, const float* y,
                                              const float* __restrict__ g,
                                              const float* __restrict__ bta,
                                              float* o) {
  const int row = blockIdx.x;
  const int tid = threadIdx.x;
  __shared__ float red[256];
  float v[3];
  float s = 0.f;
#pragma unroll
  for (int j = 0; j < 3; ++j) {
    const int i = tid + (j << 8);
    v[j] = x[(size_t)row * Dc + i] + y[(size_t)row * Dc + i];
    s += v[j];
  }
  red[tid] = s; __syncthreads();
  for (int st = 128; st > 0; st >>= 1) { if (tid < st) red[tid] += red[tid + st]; __syncthreads(); }
  const float mean = red[0] * (1.f / Dc);
  __syncthreads();
  float s2 = 0.f;
#pragma unroll
  for (int j = 0; j < 3; ++j) { const float d = v[j] - mean; s2 += d * d; }
  red[tid] = s2; __syncthreads();
  for (int st = 128; st > 0; st >>= 1) { if (tid < st) red[tid] += red[tid + st]; __syncthreads(); }
  const float inv = rsqrtf(red[0] * (1.f / Dc) + 1e-5f);
#pragma unroll
  for (int j = 0; j < 3; ++j) {
    const int i = tid + (j << 8);
    o[(size_t)row * Dc + i] = (v[j] - mean) * inv * g[i] + bta[i];
  }
}

// Self-attention, causal additive bias (-10000 above diagonal), Tk = BUF = 17.
// grid (BQ, H, BUF), block 128. qkv layout: [row][q(768) k(768) v(768)]
__global__ __launch_bounds__(128) void attn_self(const float* __restrict__ qkv,
                                                 float* __restrict__ out) {
  const int bq = blockIdx.x, h = blockIdx.y, tq = blockIdx.z;
  const int tid = threadIdx.x;
  __shared__ float sc[BUFc];
  __shared__ float invs;
  const float* qp = qkv + ((size_t)(bq * BUFc + tq)) * (3 * Dc) + h * DHc;
  if (tid < BUFc) {
    const float* kp = qkv + ((size_t)(bq * BUFc + tid)) * (3 * Dc) + Dc + h * DHc;
    float d = 0.f;
    for (int i = 0; i < DHc; ++i) d += qp[i] * kp[i];
    d *= 0.125f;                 // 1/sqrt(64)
    if (tid > tq) d -= 10000.f;  // causal additive bias (matches reference)
    sc[tid] = d;
  }
  __syncthreads();
  if (tid == 0) {
    float mx = sc[0];
    for (int i = 1; i < BUFc; ++i) mx = fmaxf(mx, sc[i]);
    float s = 0.f;
    for (int i = 0; i < BUFc; ++i) { const float e = __expf(sc[i] - mx); sc[i] = e; s += e; }
    invs = 1.f / s;
  }
  __syncthreads();
  if (tid < DHc) {
    float o = 0.f;
    for (int k = 0; k < BUFc; ++k) {
      const float* vp = qkv + ((size_t)(bq * BUFc + k)) * (3 * Dc) + 2 * Dc + h * DHc;
      o += sc[k] * vp[tid];
    }
    out[((size_t)(bq * BUFc + tq)) * Dc + h * DHc + tid] = o * invs;
  }
}

// Cross-attention over precomputed ckv[B*S][k(768) v(768)], key-padding mask.
// grid (BQ, H, BUF), block 128.
__global__ __launch_bounds__(128) void attn_cross(const float* __restrict__ qb,
                                                  const float* __restrict__ ckv,
                                                  const int* __restrict__ smask,
                                                  float* __restrict__ out) {
  const int bq = blockIdx.x, h = blockIdx.y, tq = blockIdx.z;
  const int b = bq / BEAMc;
  const int tid = threadIdx.x;
  __shared__ float sc[Sc];
  __shared__ float invs;
  const float* qp = qb + ((size_t)(bq * BUFc + tq)) * Dc + h * DHc;
  if (tid < Sc) {
    const float* kp = ckv + ((size_t)(b * Sc + tid)) * (2 * Dc) + h * DHc;
    float d = 0.f;
    for (int i = 0; i < DHc; ++i) d += qp[i] * kp[i];
    d *= 0.125f;
    if (smask[b * Sc + tid] == 0) d = -1e9f;   // where(kpm, -1e9, s)
    sc[tid] = d;
  }
  __syncthreads();
  if (tid == 0) {
    float mx = sc[0];
    for (int i = 1; i < Sc; ++i) mx = fmaxf(mx, sc[i]);
    float s = 0.f;
    for (int i = 0; i < Sc; ++i) { const float e = __expf(sc[i] - mx); sc[i] = e; s += e; }
    invs = 1.f / s;
  }
  __syncthreads();
  if (tid < DHc) {
    float o = 0.f;
    for (int k = 0; k < Sc; ++k) {
      const float* vp = ckv + ((size_t)(b * Sc + k)) * (2 * Dc) + Dc + h * DHc;
      o += sc[k] * vp[tid];
    }
    out[((size_t)(bq * BUFc + tq)) * Dc + h * DHc + tid] = o * invs;
  }
}

// h[bq,:] = x[bq, t, :]
__global__ __launch_bounds__(256) void gather_t(const float* __restrict__ x,
                                                float* __restrict__ h, int t) {
  const int bq = blockIdx.x;
  const int tid = threadIdx.x;
#pragma unroll
  for (int j = 0; j < 3; ++j) {
    const int i = tid + (j << 8);
    h[(size_t)bq * Dc + i] = x[((size_t)(bq * BUFc + t)) * Dc + i];
  }
}

// f32 -> raw bf16 (RNE), vectorized: 8 elements per iteration. n8 = n/8.
__global__ __launch_bounds__(256) void conv_bf16v(const float* __restrict__ src,
                                                  unsigned short* __restrict__ dst,
                                                  long long n8) {
  long long c = (long long)blockIdx.x * blockDim.x + threadIdx.x;
  const long long stride = (long long)gridDim.x * blockDim.x;
  for (; c < n8; c += stride) {
    const v4f f0 = *(const v4f*)(src + c * 8);
    const v4f f1 = *(const v4f*)(src + c * 8 + 4);
    v8u16 o;
#pragma unroll
    for (int i = 0; i < 4; ++i) { o[i] = f2bfu(f0[i]); o[i + 4] = f2bfu(f1[i]); }
    *(v8u16*)(dst + c * 8) = o;
  }
}

// Beam state init: buf=0 except [b][0][0]=SOS; scores = [0, NEG, NEG, NEG, NEG]
__global__ __launch_bounds__(64) void init_state(int* buf, float* scores) {
  const int tid = threadIdx.x;
  for (int i = tid; i < ROWSc; i += 64) buf[i] = 0;
  __syncthreads();
  if (tid < Bc) buf[(tid * BEAMc) * BUFc] = SOSc;
  if (tid < BQc) scores[tid] = (tid % BEAMc == 0) ? 0.f : NEGc;
}

// Per-row (20 rows) max and log-sum-exp over V for log_softmax.
__global__ __launch_bounds__(256) void rowstats(const float* __restrict__ logits,
                                                float* __restrict__ stats) {
  const int row = blockIdx.x;
  const int tid = threadIdx.x;
  __shared__ float red[256];
  const float* p = logits + (size_t)row * Vc;
  float mx = -3.4e38f;
  for (int i = tid; i < Vc; i += 256) mx = fmaxf(mx, p[i]);
  red[tid] = mx; __syncthreads();
  for (int st = 128; st > 0; st >>= 1) { if (tid < st) red[tid] = fmaxf(red[tid], red[tid + st]); __syncthreads(); }
  const float m = red[0];
  __syncthreads();
  float s = 0.f;
  for (int i = tid; i < Vc; i += 256) s += __expf(p[i] - m);
  red[tid] = s; __syncthreads();
  for (int st = 128; st > 0; st >>= 1) { if (tid < st) red[tid] += red[tid + st]; __syncthreads(); }
  if (tid == 0) { stats[row * 2] = m; stats[row * 2 + 1] = __logf(red[0]); }
}

// Per-batch top-5 over beamLk (BEAM*V), computing log_softmax + score + EOS mask
// on the fly. Updates scores in place, writes flat ids.
__global__ __launch_bounds__(256) void topk5(const float* __restrict__ logits,
                                             const float* __restrict__ stats,
                                             const int* __restrict__ buf,
                                             float* scores, int* __restrict__ ids,
                                             int t) {
  const int b = blockIdx.x;
  const int tid = threadIdx.x;
  __shared__ float sv[256 * BEAMc];
  __shared__ int   si[256 * BEAMc];
  float bvv[BEAMc]; int bii[BEAMc];
#pragma unroll
  for (int j = 0; j < BEAMc; ++j) { bvv[j] = -3.0e38f; bii[j] = 0; }
  float scs[BEAMc], mx[BEAMc], lse[BEAMc]; int last[BEAMc];
#pragma unroll
  for (int j = 0; j < BEAMc; ++j) {
    scs[j]  = scores[b * BEAMc + j];
    last[j] = buf[(b * BEAMc + j) * BUFc + t];
    mx[j]   = stats[(b * BEAMc + j) * 2];
    lse[j]  = stats[(b * BEAMc + j) * 2 + 1];
  }
  const long long total = (long long)BEAMc * Vc;
  for (long long jj = tid; jj < total; jj += 256) {
    const int beam = (int)(jj / Vc);
    const int v = (int)(jj - (long long)beam * Vc);
    float val;
    if (last[beam] == EOSc) val = NEGc;
    else val = logits[(size_t)(b * BEAMc + beam) * Vc + v] - mx[beam] - lse[beam] + scs[beam];
    if (val > bvv[BEAMc - 1]) {
      int q = BEAMc - 1;
      while (q > 0 && val > bvv[q - 1]) { bvv[q] = bvv[q - 1]; bii[q] = bii[q - 1]; --q; }
      bvv[q] = val; bii[q] = (int)jj;
    }
  }
#pragma unroll
  for (int j = 0; j < BEAMc; ++j) { sv[tid * BEAMc + j] = bvv[j]; si[tid * BEAMc + j] = bii[j]; }
  __syncthreads();
  if (tid == 0) {
    for (int s = 0; s < BEAMc; ++s) {
      float best = -3.4e38f; int bidx = 0;
      for (int i = 0; i < 256 * BEAMc; ++i)
        if (sv[i] > best) { best = sv[i]; bidx = i; }
      scores[b * BEAMc + s] = best;
      ids[b * BEAMc + s] = si[bidx];
      sv[bidx] = -3.4e38f;
    }
  }
}

// bufn[b][j][:] = bufc[b][ids/V][:]; bufn[b][j][t+1] = ids%V
__global__ __launch_bounds__(32) void reorder(const int* __restrict__ bufc,
                                              int* __restrict__ bufn,
                                              const int* __restrict__ ids, int t) {
  const int b = blockIdx.x, j = blockIdx.y, p = threadIdx.x;
  if (p >= BUFc) return;
  const int id = ids[b * BEAMc + j];
  const int prevK = id / Vc;
  const int tok = id - prevK * Vc;
  int v = bufc[(b * BEAMc + prevK) * BUFc + p];
  if (p == t + 1) v = tok;
  bufn[(b * BEAMc + j) * BUFc + p] = v;
}

// Output: scores (20 f32) then preds (20*16 i32) with cumulative EOS masking.
__global__ __launch_bounds__(32) void finalize_out(const float* __restrict__ scores,
                                                   const int* __restrict__ buf,
                                                   float* __restrict__ outs,
                                                   int* __restrict__ preds) {
  const int tid = threadIdx.x;
  if (tid < BQc) {
    outs[tid] = scores[tid];
    int cum = 0;
    for (int p = 1; p <= TMAXc; ++p) {
      const int tok = buf[tid * BUFc + p];
      if (tok == EOSc) ++cum;
      preds[tid * TMAXc + (p - 1)] = (cum == 0) ? tok : 0;
    }
  }
}

extern "C" void kernel_launch(void* const* d_in, const int* in_sizes, int n_in,
                              void* d_out, int out_size, void* d_ws, size_t ws_size,
                              hipStream_t stream) {
  (void)in_sizes; (void)n_in; (void)out_size;
  const float* emb        = (const float*)d_in[0];
  const float* pos        = (const float*)d_in[1];
  const float* ln_emb_g   = (const float*)d_in[2];
  const float* ln_emb_b   = (const float*)d_in[3];
  const float* self_in_w  = (const float*)d_in[4];
  const float* self_in_b  = (const float*)d_in[5];
  const float* self_out_w = (const float*)d_in[6];
  const float* self_out_b = (const float*)d_in[7];
  const float* cross_in_w = (const float*)d_in[8];
  const float* cross_in_b = (const float*)d_in[9];
  const float* cross_out_w= (const float*)d_in[10];
  const float* cross_out_b= (const float*)d_in[11];
  const float* ff1_w      = (const float*)d_in[12];
  const float* ff1_b      = (const float*)d_in[13];
  const float* ff2_w      = (const float*)d_in[14];
  const float* ff2_b      = (const float*)d_in[15];
  const float* ln1_g      = (const float*)d_in[16];
  const float* ln1_b      = (const float*)d_in[17];
  const float* ln2_g      = (const float*)d_in[18];
  const float* ln2_b      = (const float*)d_in[19];
  const float* ln3_g      = (const float*)d_in[20];
  const float* ln3_b      = (const float*)d_in[21];
  const float* dense_w    = (const float*)d_in[22];
  const float* dense_b    = (const float*)d_in[23];
  const float* context    = (const float*)d_in[24];
  const int*   source_mask= (const int*)d_in[25];

  // Workspace bump allocator (256B aligned)
  char* p = (char*)d_ws;
  auto alloc = [&](size_t bytes) -> void* {
    void* r = (void*)p;
    p += (bytes + 255) & ~(size_t)255;
    return r;
  };
  float* xbuf   = (float*)alloc(sizeof(float) * ROWSc * Dc);
  float* qkv    = (float*)alloc(sizeof(float) * ROWSc * 3 * Dc);
  float* qbuf   = (float*)alloc(sizeof(float) * ROWSc * Dc);
  float* att    = (float*)alloc(sizeof(float) * ROWSc * Dc);
  float* ffa    = (float*)alloc(sizeof(float) * ROWSc * DFFc);
  float* ffb    = (float*)alloc(sizeof(float) * ROWSc * Dc);
  float* ckv    = (float*)alloc(sizeof(float) * NLc * Bc * Sc * 2 * Dc);
  float* hin    = (float*)alloc(sizeof(float) * BQc * Dc);
  float* hout   = (float*)alloc(sizeof(float) * BQc * Dc);
  unsigned short* hbh = (unsigned short*)alloc(sizeof(unsigned short) * BQc * Dc);
  float* logits = (float*)alloc(sizeof(float) * BQc * Vc);
  float* stats  = (float*)alloc(sizeof(float) * BQc * 2);
  float* scores = (float*)alloc(sizeof(float) * BQc);
  int*   ids    = (int*)alloc(sizeof(int) * BQc);
  int*   bufA   = (int*)alloc(sizeof(int) * ROWSc);
  int*   bufB   = (int*)alloc(sizeof(int) * ROWSc);
  const size_t base = (size_t)(p - (char*)d_ws);
  if (base > ws_size) return;  // workspace too small: nothing sensible possible

  // bf16 embedding table (L2-resident 77MB) if workspace allows
  const size_t embh_bytes = (size_t)Vc * Dc * 2;
  unsigned short* embh = nullptr;
  if (base + embh_bytes + 256 <= ws_size) {
    embh = (unsigned short*)alloc(embh_bytes);
    conv_bf16v<<<2048, 256, 0, stream>>>(emb, embh, ((long long)Vc * Dc) / 8);
  }

  init_state<<<1, 64, 0, stream>>>(bufA, scores);

  auto gemm = [&](const float* A, const float* W, const float* bias, float* C,
                  int M, int N, int K, int act) {
    dim3 g((unsigned)((N + 15) / 16), (unsigned)((M + 15) / 16));
    if (act == 0)      gemm_wmma<0, 0, 0><<<g, 32, 0, stream>>>(A, W, bias, C, M, N, K);
    else if (act == 1) gemm_wmma<0, 0, 1><<<g, 32, 0, stream>>>(A, W, bias, C, M, N, K);
    else               gemm_wmma<0, 0, 2><<<g, 32, 0, stream>>>(A, W, bias, C, M, N, K);
  };

  // Cross-attention K/V are step-invariant: compute once per layer.
  for (int l = 0; l < NLc; ++l) {
    gemm(context,
         cross_in_w + (size_t)l * 3 * Dc * Dc + (size_t)Dc * Dc,  // rows [D, 3D)
         cross_in_b + (size_t)l * 3 * Dc + Dc,
         ckv + (size_t)l * Bc * Sc * 2 * Dc,
         Bc * Sc, 2 * Dc, Dc, 0);
  }

  int* bufc = bufA;
  int* bufn = bufB;
  for (int t = 0; t < TMAXc; ++t) {
    embed_ln<<<ROWSc, 256, 0, stream>>>(emb, pos, ln_emb_g, ln_emb_b, bufc, xbuf);
    for (int l = 0; l < NLc; ++l) {
      // self-attention block
      gemm(xbuf, self_in_w + (size_t)l * 3 * Dc * Dc, self_in_b + (size_t)l * 3 * Dc,
           qkv, ROWSc, 3 * Dc, Dc, 0);
      attn_self<<<dim3(BQc, Hc, BUFc), 128, 0, stream>>>(qkv, att);
      gemm(att, self_out_w + (size_t)l * Dc * Dc, self_out_b + (size_t)l * Dc,
           ffb, ROWSc, Dc, Dc, 0);
      add_ln<<<ROWSc, 256, 0, stream>>>(xbuf, ffb, ln1_g + (size_t)l * Dc,
                                        ln1_b + (size_t)l * Dc, xbuf);
      // cross-attention block
      gemm(xbuf, cross_in_w + (size_t)l * 3 * Dc * Dc, cross_in_b + (size_t)l * 3 * Dc,
           qbuf, ROWSc, Dc, Dc, 0);
      attn_cross<<<dim3(BQc, Hc, BUFc), 128, 0, stream>>>(
          qbuf, ckv + (size_t)l * Bc * Sc * 2 * Dc, source_mask, att);
      gemm(att, cross_out_w + (size_t)l * Dc * Dc, cross_out_b + (size_t)l * Dc,
           ffb, ROWSc, Dc, Dc, 0);
      add_ln<<<ROWSc, 256, 0, stream>>>(xbuf, ffb, ln2_g + (size_t)l * Dc,
                                        ln2_b + (size_t)l * Dc, xbuf);
      // feed-forward block
      gemm(xbuf, ff1_w + (size_t)l * DFFc * Dc, ff1_b + (size_t)l * DFFc,
           ffa, ROWSc, DFFc, Dc, 1);
      gemm(ffa, ff2_w + (size_t)l * Dc * DFFc, ff2_b + (size_t)l * Dc,
           ffb, ROWSc, Dc, DFFc, 0);
      add_ln<<<ROWSc, 256, 0, stream>>>(xbuf, ffb, ln3_g + (size_t)l * Dc,
                                        ln3_b + (size_t)l * Dc, xbuf);
    }
    // head: h = tanh(x[:,t] @ dense_w^T + dense_b); logits = h @ emb^T
    gather_t<<<BQc, 256, 0, stream>>>(xbuf, hin, t);
    gemm(hin, dense_w, dense_b, hout, BQc, Dc, Dc, 2);
    {
      dim3 g((unsigned)((Vc + 15) / 16), (unsigned)((BQc + 15) / 16));
      if (embh) {
        // pre-convert h to bf16 so the hot vocab GEMM loop is loads + WMMA only
        conv_bf16v<<<8, 256, 0, stream>>>(hout, hbh, ((long long)BQc * Dc) / 8);
        gemm_wmma<1, 1, 0><<<g, 32, 0, stream>>>(hbh, embh, nullptr, logits, BQc, Vc, Dc);
      } else {
        gemm_wmma<0, 0, 0><<<g, 32, 0, stream>>>(hout, emb, nullptr, logits, BQc, Vc, Dc);
      }
    }
    rowstats<<<BQc, 256, 0, stream>>>(logits, stats);
    topk5<<<Bc, 256, 0, stream>>>(logits, stats, bufc, scores, ids, t);
    reorder<<<dim3(Bc, BEAMc), 32, 0, stream>>>(bufc, bufn, ids, t);
    int* tmp = bufc; bufc = bufn; bufn = tmp;
  }

  finalize_out<<<1, 32, 0, stream>>>(scores, bufc, (float*)d_out,
                                     (int*)((float*)d_out) + BQc);
}
